// CharLSTM_81733227643181
// MI455X (gfx1250) — matrix operand order repaired and checked by reading, not compiled, not run on hardware
//
#include <hip/hip_runtime.h>
#include <stdint.h>

// ---------------------------------------------------------------------------
// CharLSTM forward for MI455X (gfx1250, wave32, WMMA + async-LDS copies).
//   T=8192 V=256 E=1024 H=2048, 4H=8192
// Phase 1: x = emb[seq] (bf16);  xproj = x @ Wx + b   (WMMA bf16, async LDS staging)
// Phase 2: sequential LSTM, persistent kernel, WMMA matvec:
//          broadcast-A trick (all 16 A rows = h) so D rows are all g;
//          Wh pre-packed into B-fragment tiles, streamed from L2 (33.5 MB).
//          32 WGs, ONE grid barrier per step, double-buffered gate vector,
//          h in per-WG LDS (bf16), c in registers (redundant per WG).
// Phase 3: out = hs @ Wout + bout                     (WMMA bf16)
// ---------------------------------------------------------------------------

#define TT   8192
#define VV   256
#define EE   1024
#define HH   2048
#define G4H  8192   // 4*H
#define NT   (G4H / 16)   // 512 N-tiles of 16 columns
#define KT   (HH / 32)    // 64  K-tiles of 32 rows

typedef __attribute__((ext_vector_type(16))) __bf16 v16bf;
typedef __attribute__((ext_vector_type(8)))  float  v8f;

// ---- bf16 helpers (bit-exact storage as ushort) ---------------------------
__device__ __forceinline__ unsigned short f2bf(float f) {
    unsigned int x;
    __builtin_memcpy(&x, &f, 4);
    unsigned int lsb = (x >> 16) & 1u;          // round-to-nearest-even
    x += 0x7fffu + lsb;
    return (unsigned short)(x >> 16);
}
__device__ __forceinline__ float sigmoidf_(float x) {
    return 1.0f / (1.0f + __expf(-x));
}

// ---------------------------------------------------------------------------
// Prep kernels
// ---------------------------------------------------------------------------
__global__ void init_state_kernel(unsigned* bars) {
    if (threadIdx.x < 8) bars[threadIdx.x] = 0u;   // [0]=cnt, [1]=gen
}

// x_bf16[t][e] = (bf16) emb[seq[t]][e];  one thread = 8 contiguous elems
__global__ void embed_cast_kernel(const int* __restrict__ seq,
                                  const float* __restrict__ emb,
                                  unsigned short* __restrict__ xb) {
    int i = blockIdx.x * 256 + threadIdx.x;      // chunk id, E/8=128 per row
    int t = i >> 7;
    int e = (i & 127) << 3;
    const float* src = emb + (size_t)seq[t] * EE + e;
    float4 f0 = *(const float4*)(src);
    float4 f1 = *(const float4*)(src + 4);
    unsigned int p0 = (unsigned)f2bf(f0.x) | ((unsigned)f2bf(f0.y) << 16);
    unsigned int p1 = (unsigned)f2bf(f0.z) | ((unsigned)f2bf(f0.w) << 16);
    unsigned int p2 = (unsigned)f2bf(f1.x) | ((unsigned)f2bf(f1.y) << 16);
    unsigned int p3 = (unsigned)f2bf(f1.z) | ((unsigned)f2bf(f1.w) << 16);
    *(uint4*)(xb + (size_t)i * 8) = make_uint4(p0, p1, p2, p3);
}

// WxT[n][k] = W(g)[k][j], n = g*H + j  (B-transposed for WMMA GEMM1)
__global__ void pack_wx_kernel(const float* __restrict__ Wf, const float* __restrict__ Wi,
                               const float* __restrict__ Wo, const float* __restrict__ Wc,
                               unsigned short* __restrict__ WxT) {
    int i = blockIdx.x * 256 + threadIdx.x;      // i < 4H*E
    int n = i >> 10;                              // / E
    int k = i & (EE - 1);
    int g = n >> 11;
    int j = n & (HH - 1);
    const float* W = (g == 0) ? Wf : (g == 1) ? Wi : (g == 2) ? Wo : Wc;
    WxT[(size_t)n * EE + k] = f2bf(W[(size_t)k * HH + j]);
}

// Wh packed directly into WMMA B-fragment tiles:
//   WhB[nt][kt][lane][e],  e=0..15 bf16.
//   N = nt*16 + (lane&15);  K = kt*32 + (lane&16 ? 16 : 0) + e
//   value = W(gate)[E+K][j],  gate = N>>11, j = N&2047
// A fragment lane->K layout matches (broadcast h), so one wmma per (nt,kt).
__global__ void pack_whb_kernel(const float* __restrict__ Wf, const float* __restrict__ Wi,
                                const float* __restrict__ Wo, const float* __restrict__ Wc,
                                unsigned short* __restrict__ WhB) {
    int i = blockIdx.x * 256 + threadIdx.x;      // i < NT*KT*32*16 = 2^24
    int e    = i & 15;
    int lane = (i >> 4) & 31;
    int kt   = (i >> 9) & (KT - 1);
    int nt   = i >> 15;
    int K = kt * 32 + ((lane & 16) ? 16 : 0) + e;
    int n = nt * 16 + (lane & 15);
    int g = n >> 11;
    int j = n & (HH - 1);
    const float* W = (g == 0) ? Wf : (g == 1) ? Wi : (g == 2) ? Wo : Wc;
    WhB[i] = f2bf(W[(size_t)(EE + K) * HH + j]);
}

// WoutT[n][k] = Wout[k][n]
__global__ void pack_wout_kernel(const float* __restrict__ Wout,
                                 unsigned short* __restrict__ WoutT) {
    int i = blockIdx.x * 256 + threadIdx.x;      // i < V*H
    int n = i >> 11;                              // / H
    int k = i & (HH - 1);
    WoutT[i] = f2bf(Wout[(size_t)k * VV + n]);
}

__global__ void pack_bias_kernel(const float* __restrict__ bf_, const float* __restrict__ bi_,
                                 const float* __restrict__ bo_, const float* __restrict__ bc_,
                                 float* __restrict__ bias4) {
    int n = blockIdx.x * 256 + threadIdx.x;      // n < 4H
    int g = n >> 11;
    int j = n & (HH - 1);
    const float* b = (g == 0) ? bf_ : (g == 1) ? bi_ : (g == 2) ? bo_ : bc_;
    bias4[n] = b[j];
}

// ---------------------------------------------------------------------------
// WMMA bf16 GEMM:  C[M,N] = A[M,K] * Bt[N,K]^T + bias[N]
// 128x128 tile / WG, 8 waves (4x2), each wave 32x64 = 2x4 WMMA tiles, BK=32.
// Double-buffered LDS filled with GLOBAL_LOAD_ASYNC_TO_LDS_B128 (ASYNCcnt):
// copies for tile k+1 run concurrently with the WMMAs on tile k.
// ---------------------------------------------------------------------------
__device__ __forceinline__ void stage_tile_async(unsigned short (*__restrict__ dst)[32],
                                                 const unsigned short* __restrict__ gsrc,
                                                 int ld, int tid) {
    // 128 rows x 32 bf16 = 512 chunks of 16B; 2 chunks per thread
    #pragma unroll
    for (int c = tid; c < 512; c += 256) {
        int row = c >> 2;
        int off = (c & 3) << 3;
        unsigned lds = (unsigned)(uintptr_t)(&dst[row][off]);
        unsigned long long ga =
            (unsigned long long)(uintptr_t)(gsrc + (size_t)row * ld + off);
        asm volatile("global_load_async_to_lds_b128 %0, %1, off"
                     :: "v"(lds), "v"(ga) : "memory");
    }
}
__device__ __forceinline__ void wait_async_lds() {
    asm volatile("s_wait_asynccnt 0x0" ::: "memory");
}

__global__ void __launch_bounds__(256)
gemm_bf16_wmma_kernel(const unsigned short* __restrict__ A,   // [M][K]
                      const unsigned short* __restrict__ Bt,  // [N][K]
                      const float* __restrict__ bias,         // [N]
                      float* __restrict__ C,                  // [M][N]
                      int M, int N, int K) {
    __shared__ __align__(16) unsigned short As[2][128][32];
    __shared__ __align__(16) unsigned short Bs[2][128][32];

    const int tid  = threadIdx.x;
    const int lane = tid & 31;
    const int w    = tid >> 5;
    const int wm   = w & 3;        // 0..3  -> 32-row strip
    const int wn   = w >> 2;       // 0..1  -> 64-col strip
    const int tileM = blockIdx.y * 128;
    const int tileN = blockIdx.x * 128;

    v8f acc[2][4] = {};

    const unsigned short* Ag = A  + (size_t)tileM * K;
    const unsigned short* Bg = Bt + (size_t)tileN * K;

    stage_tile_async(As[0], Ag, K, tid);
    stage_tile_async(Bs[0], Bg, K, tid);
    wait_async_lds();
    __syncthreads();

    const int nk  = K >> 5;
    const int r   = lane & 15;
    const int sel = (lane < 16);

    for (int ki = 0; ki < nk; ++ki) {
        const int cur = ki & 1;
        if (ki + 1 < nk) {
            stage_tile_async(As[cur ^ 1], Ag + ((ki + 1) << 5), K, tid);
            stage_tile_async(Bs[cur ^ 1], Bg + ((ki + 1) << 5), K, tid);
        }
        union Frag { uint4 q[2]; v16bf v; };
        Frag a[2], b[4];
        #pragma unroll
        for (int m = 0; m < 2; ++m) {
            // A 16x32: lanes 0-15: M=lane, K={0..7,16..23}; lanes 16-31: M=lane-16, K={8..15,24..31}
            const unsigned short* s = &As[cur][wm * 32 + m * 16 + r][0];
            const int ko = sel ? 0 : 8;
            a[m].q[0] = *(const uint4*)(s + ko);
            a[m].q[1] = *(const uint4*)(s + 16 + ko);
        }
        #pragma unroll
        for (int n = 0; n < 4; ++n) {
            // B 32x16 from Bt rows: lanes 0-15: N=lane, K={0..7,16..23}; lanes 16-31: N=lane-16, K={8..15,24..31}
            const unsigned short* s = &Bs[cur][wn * 64 + n * 16 + r][0];
            const int ko = sel ? 0 : 8;
            b[n].q[0] = *(const uint4*)(s + ko);
            b[n].q[1] = *(const uint4*)(s + 16 + ko);
        }
        #pragma unroll
        for (int m = 0; m < 2; ++m)
            #pragma unroll
            for (int n = 0; n < 4; ++n)
                acc[m][n] = __builtin_amdgcn_wmma_f32_16x16x32_bf16(
                    false, a[m].v, false, b[n].v, (short)0, acc[m][n], false, false);
        if (ki + 1 < nk) wait_async_lds();
        __syncthreads();
    }

    #pragma unroll
    for (int m = 0; m < 2; ++m) {
        #pragma unroll
        for (int n = 0; n < 4; ++n) {
            const int colBase = tileN + wn * 64 + n * 16 + (lane & 15);
            const int rowBase = tileM + wm * 32 + m * 16 + (sel ? 0 : 8);
            const float bb = bias[colBase];
            #pragma unroll
            for (int rr = 0; rr < 8; ++rr) {
                C[(size_t)(rowBase + rr) * N + colBase] = acc[m][n][rr] + bb;
            }
        }
    }
}

// ---------------------------------------------------------------------------
// Persistent LSTM recurrence with WMMA matvec.
//   32 WGs x 8 waves = 256 waves; wave gw owns N-tiles {2gw, 2gw+1}.
//   A fragment: broadcast h (every lane loads the K-slice its lane-group
//   needs, independent of M) -> every D row equals g. Two independent
//   64-step accumulation chains per wave hide XDL latency.
//   One grid barrier per step; gate vector double-buffered over t&1.
// ---------------------------------------------------------------------------
#define RWG 32

__device__ __forceinline__ void grid_barrier(unsigned* cnt, unsigned* gen) {
    __threadfence();
    __syncthreads();
    if (threadIdx.x == 0) {
        unsigned g = __hip_atomic_load(gen, __ATOMIC_ACQUIRE, __HIP_MEMORY_SCOPE_AGENT);
        unsigned arrived = __hip_atomic_fetch_add(cnt, 1u, __ATOMIC_ACQ_REL, __HIP_MEMORY_SCOPE_AGENT);
        if (arrived == RWG - 1u) {
            __hip_atomic_store(cnt, 0u, __ATOMIC_RELAXED, __HIP_MEMORY_SCOPE_AGENT);
            __hip_atomic_fetch_add(gen, 1u, __ATOMIC_RELEASE, __HIP_MEMORY_SCOPE_AGENT);
        } else {
            while (__hip_atomic_load(gen, __ATOMIC_ACQUIRE, __HIP_MEMORY_SCOPE_AGENT) == g) {
                __builtin_amdgcn_s_sleep(1);
            }
        }
    }
    __syncthreads();
}

__global__ void __launch_bounds__(256)
lstm_recurrence_kernel(const float* __restrict__ xproj,        // [T][4H]
                       const unsigned short* __restrict__ WhB, // [NT][KT][32][16] bf16
                       float* __restrict__ pbuf,               // [2][4H] gate vector
                       unsigned short* __restrict__ hs,        // [T][H] bf16
                       float* __restrict__ hc_out,             // h[H] then c[H]
                       unsigned* __restrict__ bars) {
    __shared__ __align__(16) unsigned short hsm_bf[HH];   // this WG's h (bf16)
    const int tid  = threadIdx.x;
    const int wg   = blockIdx.x;
    const int lane = tid & 31;
    const int w    = tid >> 5;
    const int gw   = wg * 8 + w;        // global wave id 0..255
    const int nt0  = gw * 2;            // two 16-column N-tiles per wave
    unsigned* cnt = bars;
    unsigned* gen = bars + 1;

    float c8[8], h8[8];                 // cell/hidden state (redundant per WG)
    #pragma unroll
    for (int q = 0; q < 8; ++q) { c8[q] = 0.0f; h8[q] = 0.0f; }
    for (int i = tid; i < HH; i += 256) hsm_bf[i] = 0;
    __syncthreads();

    for (int t = 0; t < TT; ++t) {
        // ---- matvec on the matrix unit: g = h @ Wh (+ xproj) ----
        union Frag { uint4 q[2]; v16bf v; };
        v8f acc0 = {}, acc1 = {};
        const unsigned short* wb0 = WhB + (size_t)(nt0 + 0) * KT * 32 * 16 + lane * 16;
        const unsigned short* wb1 = WhB + (size_t)(nt0 + 1) * KT * 32 * 16 + lane * 16;
        for (int kt = 0; kt < KT; ++kt) {
            Frag af, b0, b1;
            // broadcast-A: lanes 0-15 load K=0..15, lanes 16-31 K=16..31
            const unsigned short* hp = &hsm_bf[kt * 32 + ((lane & 16) ? 16 : 0)];
            af.q[0] = *(const uint4*)(hp);
            af.q[1] = *(const uint4*)(hp + 8);
            const uint4* p0 = (const uint4*)(wb0 + (size_t)kt * 32 * 16);
            b0.q[0] = p0[0]; b0.q[1] = p0[1];
            const uint4* p1 = (const uint4*)(wb1 + (size_t)kt * 32 * 16);
            b1.q[0] = p1[0]; b1.q[1] = p1[1];
            acc0 = __builtin_amdgcn_wmma_f32_16x16x32_bf16(
                false, af.v, false, b0.v, (short)0, acc0, false, false);
            acc1 = __builtin_amdgcn_wmma_f32_16x16x32_bf16(
                false, af.v, false, b1.v, (short)0, acc1, false, false);
        }
        float* pb = pbuf + (size_t)(t & 1) * G4H;
        if (lane < 16) {                 // row 0 of D (all rows identical)
            const int n0 = nt0 * 16 + lane;
            pb[n0]      = acc0[0] + xproj[(size_t)t * G4H + n0];
            pb[n0 + 16] = acc1[0] + xproj[(size_t)t * G4H + n0 + 16];
        }

        grid_barrier(cnt, gen);          // the ONLY grid sync per step

        // ---- combine: thread owns j = tid*8 .. tid*8+7 (redundant per WG) ----
        const float* pbr = pbuf + (size_t)(t & 1) * G4H;
        float gacc[4][8];
        #pragma unroll
        for (int g = 0; g < 4; ++g) {
            const float4* p = (const float4*)(pbr + g * HH + tid * 8);
            float4 u0 = p[0], u1 = p[1];
            gacc[g][0] = u0.x; gacc[g][1] = u0.y; gacc[g][2] = u0.z; gacc[g][3] = u0.w;
            gacc[g][4] = u1.x; gacc[g][5] = u1.y; gacc[g][6] = u1.z; gacc[g][7] = u1.w;
        }
        #pragma unroll
        for (int q = 0; q < 8; ++q) {
            const int j  = tid * 8 + q;
            const float f  = sigmoidf_(gacc[0][q]);
            const float i_ = sigmoidf_(gacc[1][q]);
            const float o  = sigmoidf_(gacc[2][q]);
            const float cd = tanhf(gacc[3][q]);
            c8[q] = f * c8[q] + i_ * cd;
            const float h = o * tanhf(c8[q]);
            h8[q] = h;
            const unsigned short hb = f2bf(h);
            hsm_bf[j] = hb;
            if (wg == 0) hs[(size_t)t * HH + j] = hb;
        }
        __syncthreads();                 // hsm_bf ready for next matvec
    }

    if (wg == 0) {
        #pragma unroll
        for (int q = 0; q < 8; ++q) {
            const int j = tid * 8 + q;
            hc_out[j]      = h8[q];    // final h (f32)
            hc_out[HH + j] = c8[q];    // final c (f32)
        }
    }
}

// ---------------------------------------------------------------------------
// Host launcher
// ---------------------------------------------------------------------------
extern "C" void kernel_launch(void* const* d_in, const int* in_sizes, int n_in,
                              void* d_out, int out_size, void* d_ws, size_t ws_size,
                              hipStream_t stream) {
    (void)in_sizes; (void)n_in; (void)out_size; (void)ws_size;

    const int*   seq  = (const int*)  d_in[0];
    const float* emb  = (const float*)d_in[1];
    const float* Wf   = (const float*)d_in[2];
    const float* bf_  = (const float*)d_in[3];
    const float* Wi   = (const float*)d_in[4];
    const float* bi_  = (const float*)d_in[5];
    const float* Wo   = (const float*)d_in[6];
    const float* bo_  = (const float*)d_in[7];
    const float* Wc   = (const float*)d_in[8];
    const float* bc_  = (const float*)d_in[9];
    const float* Wout = (const float*)d_in[10];
    const float* bout = (const float*)d_in[11];

    uint8_t* ws = (uint8_t*)d_ws;
    size_t off = 0;
    auto take = [&](size_t bytes) -> void* {
        void* p = ws + off;
        off = (off + bytes + 255) & ~(size_t)255;
        return p;
    };

    unsigned short* x_bf  = (unsigned short*)take((size_t)TT * EE * 2);     // 16 MB
    unsigned short* WxT   = (unsigned short*)take((size_t)G4H * EE * 2);    // 16 MB
    unsigned short* WhB   = (unsigned short*)take((size_t)NT * KT * 32 * 16 * 2); // 32 MB (L2)
    unsigned short* WoutT = (unsigned short*)take((size_t)VV * HH * 2);     // 1 MB
    float*          bias4 = (float*)take((size_t)G4H * 4);
    float*          xproj = (float*)take((size_t)TT * G4H * 4);             // 256 MB
    unsigned short* hsbuf = (unsigned short*)take((size_t)TT * HH * 2);     // 32 MB
    float*          pbuf  = (float*)take((size_t)2 * G4H * 4);              // 64 KB
    unsigned*       bars  = (unsigned*)take(256);

    float* out    = (float*)d_out;           // [T][V]
    float* hc_out = out + (size_t)TT * VV;   // h[H] then c[H]

    // --- prep ---
    init_state_kernel<<<1, 256, 0, stream>>>(bars);
    embed_cast_kernel<<<(TT * (EE / 8)) / 256, 256, 0, stream>>>(seq, emb, x_bf);
    pack_wx_kernel <<<((size_t)G4H * EE) / 256, 256, 0, stream>>>(Wf, Wi, Wo, Wc, WxT);
    pack_whb_kernel<<<((size_t)NT * KT * 32 * 16) / 256, 256, 0, stream>>>(Wf, Wi, Wo, Wc, WhB);
    pack_wout_kernel<<<((size_t)VV * HH) / 256, 256, 0, stream>>>(Wout, WoutT);
    pack_bias_kernel<<<G4H / 256, 256, 0, stream>>>(bf_, bi_, bo_, bc_, bias4);

    // --- phase 1: xproj = x @ Wx + b  (M=8192, N=8192, K=1024) ---
    gemm_bf16_wmma_kernel<<<dim3(G4H / 128, TT / 128), 256, 0, stream>>>(
        x_bf, WxT, bias4, xproj, TT, G4H, EE);

    // --- phase 2: sequential recurrence (32 WGs, WMMA matvec, 1 barrier/step) ---
    lstm_recurrence_kernel<<<RWG, 256, 0, stream>>>(
        xproj, WhB, pbuf, hsbuf, hc_out, bars);

    // --- phase 3: out = hs @ Wout + bout  (M=8192, N=256, K=2048) ---
    gemm_bf16_wmma_kernel<<<dim3(VV / 128, TT / 128), 256, 0, stream>>>(
        hsbuf, WoutT, bout, out, TT, VV, HH);
}
